// ConvNetLayer_24824910970967
// MI455X (gfx1250) — matrix operand order repaired and checked by reading, not compiled, not run on hardware
//
#include <hip/hip_runtime.h>

typedef _Float16 v16h __attribute__((ext_vector_type(16)));
typedef _Float16 v8h  __attribute__((ext_vector_type(8)));
typedef _Float16 v4h  __attribute__((ext_vector_type(4)));
typedef float    v8f  __attribute__((ext_vector_type(8)));
typedef float    v4f  __attribute__((ext_vector_type(4)));
typedef int      v4i  __attribute__((ext_vector_type(4)));

#define GN 8192   // nodes
#define GD 512    // node dim
#define TI 128    // kernel1: output rows per workgroup
#define KJ 32     // K step
#define APDW 18   // ldsA pitch in dwords (= 36 f16)
#define PITCH 40  // kernel2 LDS pitch in f16

static __device__ __forceinline__ v8h cat8(v4h a, v4h b) {
  return __builtin_shufflevector(a, b, 0,1,2,3,4,5,6,7);
}
static __device__ __forceinline__ v16h cat16(v8h a, v8h b) {
  return __builtin_shufflevector(a, b, 0,1,2,3,4,5,6,7,8,9,10,11,12,13,14,15);
}

// ---------------------------------------------------------------------------
// Kernel 0: build xT = x^T as f16, layout [d][j] (GD x GN).  One-time 24 MB.
// ---------------------------------------------------------------------------
__global__ void xT_kernel(const float* __restrict__ x, _Float16* __restrict__ xT) {
  const int g  = blockIdx.x * 256 + threadIdx.x;
  const int d4 = (g & 127) << 2;   // 0..508
  const int j0 = (g >> 7) << 2;    // 0..8188
  const v4f r0 = *(const v4f*)(x + (size_t)(j0 + 0) * GD + d4);
  const v4f r1 = *(const v4f*)(x + (size_t)(j0 + 1) * GD + d4);
  const v4f r2 = *(const v4f*)(x + (size_t)(j0 + 2) * GD + d4);
  const v4f r3 = *(const v4f*)(x + (size_t)(j0 + 3) * GD + d4);
  #pragma unroll
  for (int e = 0; e < 4; e++) {
    const v4h o = { (_Float16)r0[e], (_Float16)r1[e], (_Float16)r2[e], (_Float16)r3[e] };
    *(v4h*)(xT + (size_t)(d4 + e) * GN + j0) = o;
  }
}

// ---------------------------------------------------------------------------
// Kernel 1: agg[i,d] = (sum_j adj[j,i] * x[j,d]) / deg[i]  ->  f16 aggh.
// WG = 128 i x all 512 d; 16 waves (2 wi x 8 wd); wave = 4x4 WMMA tiles.
// adj tile double-pumped through registers (2-deep pipeline) to hide HBM
// latency; B fragments direct from L2-resident xT with prefetch.
// ---------------------------------------------------------------------------
__global__ __launch_bounds__(512) void gcn_agg_kernel(
    const _Float16* __restrict__ xT, const int* __restrict__ adj,
    _Float16* __restrict__ aggh) {
  __shared__ __align__(16) unsigned int ldsA[TI * APDW];  // packed f16x2: [i][j-pair]
  __shared__ float ldsDeg[TI];

  const int t    = threadIdx.x;
  const int lane = t & 31;
  const int w    = t >> 5;
  const int wi   = w >> 3;   // 0..1  -> 64 i-rows
  const int wd   = w & 7;    // 0..7  -> 64 d-cols
  const int ib   = blockIdx.x * TI;

  if (t < TI) ldsDeg[t] = 0.0f;

  // adj staging: thread covers i = 4*(t&31)..+3, j = {2*(t>>5), +1}
  const int ai  = (t & 31) << 2;   // 0..124
  const int ajp = t >> 5;          // 0..15 (j dword-pair)
  float dg0 = 0.f, dg1 = 0.f, dg2 = 0.f, dg3 = 0.f;

  const int adw0 = (wi * 64 + (lane & 15)) * APDW + ((lane < 16) ? 0 : 4);
  const size_t bbase = (size_t)(wd * 64 + (lane & 15)) * GN + ((lane >> 4) << 3);
  const _Float16* bp = xT + bbase;

  v8f acc[4][4] = {};

  const int* adjRow0 = adj + (size_t)(2 * ajp) * GN + ib + ai;      // j = 2*ajp
  const int* adjRow1 = adj + (size_t)(2 * ajp + 1) * GN + ib + ai;  // j = 2*ajp+1

  auto ldTile = [&](int jb, v4i& a0, v4i& a1) {
    a0 = __builtin_nontemporal_load((const v4i*)(adjRow0 + (size_t)jb * GN));
    a1 = __builtin_nontemporal_load((const v4i*)(adjRow1 + (size_t)jb * GN));
  };

  auto stage = [&](const v4i& a0, const v4i& a1) {
    #pragma unroll
    for (int c = 0; c < 4; c++) {
      union { v4h h; unsigned int u[2]; } pk;
      pk.h = (v4h){ (_Float16)(float)a0[c], (_Float16)(float)a1[c],
                    (_Float16)0.f, (_Float16)0.f };
      ldsA[(ai + c) * APDW + ajp] = pk.u[0];
    }
    dg0 += (float)(a0[0] + a1[0]);
    dg1 += (float)(a0[1] + a1[1]);
    dg2 += (float)(a0[2] + a1[2]);
    dg3 += (float)(a0[3] + a1[3]);
  };

  auto compute = [&](int jb) {
    // prefetch next iteration's B fragments (counter-free global_prefetch)
    #pragma unroll
    for (int dt = 0; dt < 4; dt++)
      __builtin_prefetch(bp + (size_t)dt * 16 * GN + (jb + KJ), 0, 3);

    v16h afr[4];
    #pragma unroll
    for (int it = 0; it < 4; it++) {
      const unsigned int* ap = &ldsA[adw0 + it * 16 * APDW];
      const v4h q0 = *(const v4h*)(ap + 0);
      const v4h q1 = *(const v4h*)(ap + 2);
      const v4h q2 = *(const v4h*)(ap + 8);
      const v4h q3 = *(const v4h*)(ap + 10);
      afr[it] = cat16(cat8(q0, q1), cat8(q2, q3));
    }
    const _Float16* p = bp + jb;
    #pragma unroll
    for (int dt = 0; dt < 4; dt++) {
      const v8h lo = *(const v8h*)(p + (size_t)dt * 16 * GN);
      const v8h hi = *(const v8h*)(p + (size_t)dt * 16 * GN + 16);
      const v16h bfr = cat16(lo, hi);
      #pragma unroll
      for (int it = 0; it < 4; it++)
        acc[it][dt] = __builtin_amdgcn_wmma_f32_16x16x32_f16(
            false, afr[it], false, bfr, (short)0, acc[it][dt], false, false);
    }
  };

  // --- 2-deep software pipeline over adj tiles ---
  v4i p0a, p0b, p1a, p1b;
  ldTile(0, p0a, p0b);
  ldTile(KJ, p1a, p1b);

  for (int jb = 0; jb < GN; jb += 2 * KJ) {
    stage(p0a, p0b);
    __syncthreads();
    if (jb + 2 * KJ < GN) ldTile(jb + 2 * KJ, p0a, p0b);
    compute(jb);
    __syncthreads();

    stage(p1a, p1b);
    __syncthreads();
    if (jb + 3 * KJ < GN) ldTile(jb + 3 * KJ, p1a, p1b);
    compute(jb + KJ);
    __syncthreads();
  }

  // --- deg reduce + invert ---
  atomicAdd(&ldsDeg[ai + 0], dg0);
  atomicAdd(&ldsDeg[ai + 1], dg1);
  atomicAdd(&ldsDeg[ai + 2], dg2);
  atomicAdd(&ldsDeg[ai + 3], dg3);
  __syncthreads();
  if (t < TI) {
    const float dv = ldsDeg[t];
    ldsDeg[t] = dv > 0.f ? 1.f / dv : 0.f;
  }
  __syncthreads();

  // --- epilogue: scale by 1/deg, store f16 agg ---
  #pragma unroll
  for (int it = 0; it < 4; it++) {
    const int irow = wi * 64 + it * 16 + ((lane >> 4) << 3);
    #pragma unroll
    for (int p = 0; p < 8; p++) {
      const float rd = ldsDeg[irow + p];
      const size_t gi = (size_t)(ib + irow + p) * GD;
      #pragma unroll
      for (int dt = 0; dt < 4; dt++) {
        const int dgc = wd * 64 + dt * 16 + (lane & 15);
        aggh[gi + dgc] = (_Float16)(acc[it][dt][p] * rd);
      }
    }
  }
}

// ---------------------------------------------------------------------------
// Kernel 2: out = relu(agg @ U^T).  A (agg f16 row-major) direct from global;
// U^T tiles staged to LDS as f16 [k][d_local].
// ---------------------------------------------------------------------------
__global__ __launch_bounds__(512) void gcn_out_kernel(
    const _Float16* __restrict__ aggh, const float* __restrict__ U,
    float* __restrict__ out) {
  __shared__ __align__(16) _Float16 ldsU[GD * PITCH];   // [k][d_local]

  const int t    = threadIdx.x;
  const int lane = t & 31;
  const int w    = t >> 5;
  const int wi   = w >> 3;   // 0..1
  const int wk   = w & 7;    // 0..7
  const int ib   = blockIdx.x * 64;

  const int bcol = (wk * 64 + (lane & 15)) * PITCH;
  const int boff = (lane >> 4) << 3;
  const int aoff = (lane < 16) ? 0 : 8;
  const int arow = ib + wi * 32 + (lane & 15);

  v8f acc[2][4] = {};

  for (int db = 0; db < GD; db += KJ) {
    __syncthreads();

    #pragma unroll
    for (int r = 0; r < 8; r++) {
      const int fi = r * 512 + t;
      const int k  = fi >> 3;
      const int c4 = (fi & 7) << 2;
      const v4f uv = *(const v4f*)(U + (size_t)k * GD + db + c4);
      const v4h uh = { (_Float16)uv[0], (_Float16)uv[1], (_Float16)uv[2], (_Float16)uv[3] };
      *(v4h*)&ldsU[k * PITCH + c4] = uh;
    }

    __syncthreads();

    v16h afr[2];
    #pragma unroll
    for (int it = 0; it < 2; it++) {
      const _Float16* ap = aggh + (size_t)(arow + it * 16) * GD + db + aoff;
      const v8h lo = *(const v8h*)ap;
      const v8h hi = *(const v8h*)(ap + 16);
      afr[it] = cat16(lo, hi);
    }
    #pragma unroll
    for (int dt = 0; dt < 4; dt++) {
      const v8h lo = *(const v8h*)&ldsU[bcol + dt * 16 * PITCH + boff];
      const v8h hi = *(const v8h*)&ldsU[bcol + dt * 16 * PITCH + boff + 16];
      const v16h bfr = cat16(lo, hi);
      #pragma unroll
      for (int it = 0; it < 2; it++)
        acc[it][dt] = __builtin_amdgcn_wmma_f32_16x16x32_f16(
            false, afr[it], false, bfr, (short)0, acc[it][dt], false, false);
    }
  }

  #pragma unroll
  for (int it = 0; it < 2; it++) {
    const int irow = wi * 32 + it * 16 + ((lane >> 4) << 3);
    #pragma unroll
    for (int p = 0; p < 8; p++) {
      const size_t go = (size_t)(ib + irow + p) * GD;
      #pragma unroll
      for (int dt = 0; dt < 4; dt++) {
        const int kg = wk * 64 + dt * 16 + (lane & 15);
        const float v = acc[it][dt][p];
        out[go + kg] = v > 0.f ? v : 0.f;
      }
    }
  }
}

// ---------------------------------------------------------------------------
extern "C" void kernel_launch(void* const* d_in, const int* in_sizes, int n_in,
                              void* d_out, int out_size, void* d_ws, size_t ws_size,
                              hipStream_t stream) {
  const float* x   = (const float*)d_in[0];   // [N, D] f32
  const int*   adj = (const int*)d_in[1];     // [N, N] i32 (0/1)
  const float* U   = (const float*)d_in[2];   // [D, D] f32
  float* out = (float*)d_out;                 // [N, D] f32

  _Float16* xTh  = (_Float16*)d_ws;                 // 8 MB: x^T as f16 [d][j]
  _Float16* aggh = xTh + (size_t)GN * GD;           // 8 MB: agg as f16 [i][d]

  xT_kernel<<<(GN / 4) * (GD / 4) / 256, 256, 0, stream>>>(x, xTh);
  gcn_agg_kernel<<<GN / TI, 512, 0, stream>>>(xTh, adj, aggh);
  gcn_out_kernel<<<GN / 64, 512, 0, stream>>>(aggh, U, out);
}
